// LSTM_73366631350908
// MI455X (gfx1250) — matrix operand order repaired and checked by reading, not compiled, not run on hardware
//
#include <hip/hip_runtime.h>
#include <hip/hip_bf16.h>

#define HID     1024
#define BATCH   64
#define SRC_LEN 1024
#define TGT_LEN 512
#define IN_DIM  128
#define OUT_DIM 128
#define NBLK    64
#define GATES   4
#define NTHR    256                           // 8 waves: (gp 0..1) x (rowtile 0..3)

// LDS: Whh0 slab, Wih1 slab (each 4 gates x 16 rows x 1024 k bf16), then f32 scratch
#define SLAB0 0
#define SLAB1 (GATES * 16 * HID)
#define LDS_ELEMS (2 * GATES * 16 * HID)      // 131072 bf16 = 256 KB
#define SCRATCH_F32 2048                      // 8 tiles x 32 lanes x 8 f32 = 8 KB

typedef __bf16 bf16_t;
typedef __attribute__((ext_vector_type(16))) __bf16 v16bf;
typedef __attribute__((ext_vector_type(8)))  __bf16 v8bf;
typedef __attribute__((ext_vector_type(8)))  float  v8f;
typedef unsigned int u32x4 __attribute__((ext_vector_type(4)));
typedef int          i32x4 __attribute__((ext_vector_type(4)));
typedef int          i32x8 __attribute__((ext_vector_type(8)));

#if defined(__gfx1250__) && __has_builtin(__builtin_amdgcn_tensor_load_to_lds)
#define HAVE_TDM 1
#else
#define HAVE_TDM 0
#endif

extern __shared__ bf16_t lds_w[];

__device__ __forceinline__ v16bf cat8(v8bf lo, v8bf hi) {
  return __builtin_shufflevector(lo, hi, 0,1,2,3,4,5,6,7,8,9,10,11,12,13,14,15);
}

// A tile (16 rows x 32 k), row-major source with row stride lda.
// A as loaded here (contiguous k, B matches): lane<16: m=lane, K={0..15}; lane>=16: m=lane-16, K={16..31}
__device__ __forceinline__ v16bf load_a16(const bf16_t* __restrict__ p, int lda, int lane) {
  const int m  = lane & 15;
  const int kb = (lane >> 4) << 4;
  const v8bf* q = (const v8bf*)(p + (size_t)m * lda + kb);
  return cat8(q[0], q[1]);
}

// B tile (32 k x 16 cols): B[k][n] = W[n][k], W row-major over k (stride ldw), global.
__device__ __forceinline__ v16bf load_b16(const bf16_t* __restrict__ p, int ldw, int lane) {
  const int nn = lane & 15;
  const int kb = (lane >> 4) << 4;
  const v8bf* q = (const v8bf*)(p + (size_t)nn * ldw + kb);
  return cat8(q[0], q[1]);
}

// Same B layout, from the LDS weight slabs (row stride fixed = HID).
__device__ __forceinline__ v16bf load_b16_lds(unsigned base, int lane) {
  const int nn = lane & 15;
  const int kb = (lane >> 4) << 4;
  const v8bf* q = (const v8bf*)&lds_w[base + (unsigned)nn * HID + (unsigned)kb];
  return cat8(q[0], q[1]);
}

__device__ __forceinline__ v8f wmma_bf16(v16bf a, v16bf b, v8f c) {
  return __builtin_amdgcn_wmma_f32_16x16x32_bf16(false, a, false, b, (short)0, c,
                                                 false, false);
}

__device__ __forceinline__ float sigf(float x) { return 1.f / (1.f + __expf(-x)); }

__device__ __forceinline__ void grid_barrier(unsigned* cnt, unsigned target) {
  __threadfence();
  __syncthreads();
  if (threadIdx.x == 0) {
    atomicAdd(cnt, 1u);
    while (__hip_atomic_load(cnt, __ATOMIC_ACQUIRE, __HIP_MEMORY_SCOPE_AGENT) < target) {
      __builtin_amdgcn_s_sleep(1);
    }
  }
  __syncthreads();
}

#if HAVE_TDM
// One TDM 2D tile: 16 rows x HID bf16, row stride HID elems, global -> LDS byte offset.
__device__ __forceinline__ void tdm_load_tile(const bf16_t* gsrc, unsigned lds_byte_off) {
  unsigned long long ga = (unsigned long long)gsrc;
  u32x4 g0 = { 1u,                                   // count=1 (valid user descriptor)
               lds_byte_off,                         // lds_addr (bytes)
               (unsigned)(ga & 0xffffffffu),         // global_addr[31:0]
               ((unsigned)((ga >> 32) & 0x01ffffffu)) | 0x80000000u };  // addr[56:32]|type=2
  i32x8 g1 = { (int)0x00010000u,                     // data_size=1 (2 bytes)
               (int)(((unsigned)HID & 0xffffu) << 16),   // tensor_dim0 lo -> [31:16]
               (int)(16u << 16),                     // tensor_dim0 hi=0 | tensor_dim1=16
               (int)(((unsigned)HID & 0xffffu) << 16),   // tensor_dim1 hi=0 | tile_dim0=HID
               16,                                   // tile_dim1=16, tile_dim2=0
               HID,                                  // tensor_dim0_stride[31:0]
               0, 0 };
  i32x4 gz = { 0, 0, 0, 0 };
#if defined(__clang_major__) && (__clang_major__ >= 23)
  i32x8 gz8 = { 0, 0, 0, 0, 0, 0, 0, 0 };
  __builtin_amdgcn_tensor_load_to_lds(g0, g1, gz, gz, gz8, 0);
#else
  __builtin_amdgcn_tensor_load_to_lds(g0, g1, gz, gz, 0);
#endif
}
#endif

// Stage the block's time-invariant Whh0 / Wih1 column slices into LDS (once).
__device__ __forceinline__ void stage_weights(const bf16_t* __restrict__ Whh0,
                                              const bf16_t* __restrict__ Wih1,
                                              int c0, int tid) {
#if HAVE_TDM
  if (tid < 32) {  // TDM is a wave-level DMA op; one wave issues all descriptors
#pragma unroll
    for (int g = 0; g < GATES; ++g) {
      tdm_load_tile(Whh0 + (size_t)(g * HID + c0) * HID,
                    (unsigned)((SLAB0 + g * 16 * HID) * 2));
      tdm_load_tile(Wih1 + (size_t)(g * HID + c0) * HID,
                    (unsigned)((SLAB1 + g * 16 * HID) * 2));
    }
    __builtin_amdgcn_s_wait_tensorcnt(0);
  }
#else
  const int CH = HID / 8;                       // 16B chunks per row
  for (int idx = tid; idx < GATES * 16 * CH; idx += NTHR) {
    const int ch  = idx % CH;
    const int row = idx / CH;                   // g*16 + r
    const int g   = row >> 4, r = row & 15;
    v8bf v0 = *(const v8bf*)(Whh0 + (size_t)(g * HID + c0 + r) * HID + ch * 8);
    v8bf v1 = *(const v8bf*)(Wih1 + (size_t)(g * HID + c0 + r) * HID + ch * 8);
    *(v8bf*)&lds_w[SLAB0 + row * HID + ch * 8] = v0;
    *(v8bf*)&lds_w[SLAB1 + row * HID + ch * 8] = v1;
  }
#endif
  __syncthreads();
}

// acc[0..1] += A(16xK) x {W0,W1}(16xK)^T, weights global; software-pipelined.
__device__ __forceinline__ void mma2_glb(v8f acc[2],
    const bf16_t* __restrict__ A, int lda,
    const bf16_t* __restrict__ W0, const bf16_t* __restrict__ W1, int ldw,
    int K, int lane) {
  v16bf a  = load_a16(A, lda, lane);
  v16bf b0 = load_b16(W0, ldw, lane);
  v16bf b1 = load_b16(W1, ldw, lane);
#pragma unroll 4
  for (int k0 = 0; k0 < K; k0 += 32) {
    v16bf an = a, bn0 = b0, bn1 = b1;
    if (k0 + 32 < K) {                 // prefetch next iteration before issuing WMMAs
      an  = load_a16(A + k0 + 32, lda, lane);
      bn0 = load_b16(W0 + k0 + 32, ldw, lane);
      bn1 = load_b16(W1 + k0 + 32, ldw, lane);
    }
    acc[0] = wmma_bf16(a, b0, acc[0]);
    acc[1] = wmma_bf16(a, b1, acc[1]);
    a = an; b0 = bn0; b1 = bn1;
  }
}

// acc[0..1] += A(16x1024) x LDS slab tiles; software-pipelined.
__device__ __forceinline__ void mma2_lds(v8f acc[2],
    const bf16_t* __restrict__ A, int lda,
    unsigned base0, unsigned base1, int lane) {
  v16bf a  = load_a16(A, lda, lane);
  v16bf b0 = load_b16_lds(base0, lane);
  v16bf b1 = load_b16_lds(base1, lane);
#pragma unroll 4
  for (int k0 = 0; k0 < HID; k0 += 32) {
    v16bf an = a, bn0 = b0, bn1 = b1;
    if (k0 + 32 < HID) {
      an  = load_a16(A + k0 + 32, lda, lane);
      bn0 = load_b16_lds(base0 + (unsigned)k0 + 32u, lane);
      bn1 = load_b16_lds(base1 + (unsigned)k0 + 32u, lane);
    }
    acc[0] = wmma_bf16(a, b0, acc[0]);
    acc[1] = wmma_bf16(a, b1, acc[1]);
    a = an; b0 = bn0; b1 = bn1;
  }
}

__global__ void cvt_f32_bf16(const float* __restrict__ in, bf16_t* __restrict__ out,
                             long long n) {
  long long i = (long long)blockIdx.x * blockDim.x + threadIdx.x;
  const long long stride = (long long)gridDim.x * blockDim.x;
  for (; i < n; i += stride) out[i] = (bf16_t)in[i];
}

__global__ void add_bias(const float* __restrict__ a, const float* __restrict__ b,
                         float* __restrict__ o, int n) {
  int i = blockIdx.x * blockDim.x + threadIdx.x;
  if (i < n) o[i] = a[i] + b[i];
}

__global__ __launch_bounds__(NTHR) void lstm_seq2seq(
    const bf16_t* __restrict__ srcb, const bf16_t* __restrict__ tgtb,
    const bf16_t* __restrict__ Wih0, const bf16_t* __restrict__ Whh0,
    const bf16_t* __restrict__ Wih1, const bf16_t* __restrict__ Whh1,
    const bf16_t* __restrict__ Wfc,
    const float* __restrict__ b0, const float* __restrict__ b1,
    const float* __restrict__ bfc,
    bf16_t* __restrict__ h0buf, bf16_t* __restrict__ h1buf,
    float* __restrict__ out, unsigned* __restrict__ sync)
{
  const int tid     = threadIdx.x;
  const int wave    = tid >> 5;
  const int rt      = wave & 3;          // batch row-tile 0..3
  const int gp      = wave >> 2;         // 0: gates (i,f)   1: gates (g,o)
  const int lane    = tid & 31;
  const int nn      = lane & 15;
  const int hi      = lane >> 4;
  const int blk     = blockIdx.x;        // hidden cols [blk*16, blk*16+16)
  const int c0      = blk * 16;
  const int rowbase = rt * 16;
  const int g0      = gp * 2, g1 = gp * 2 + 1;

  float* scratch = (float*)&lds_w[LDS_ELEMS];       // 8 KB f32 gate-exchange

  // One-time: DMA this block's Whh0 / Wih1 column slices into LDS via TDM.
  stage_weights(Whh0, Whh1 /*unused arg kept uniform*/ == Whh1 ? Wih1 : Wih1, c0, tid);

  float bias0[GATES], bias1[GATES];
#pragma unroll
  for (int g = 0; g < GATES; ++g) {
    bias0[g] = b0[g * HID + c0 + nn];
    bias1[g] = b1[g * HID + c0 + nn];
  }

  // Cell state lives in VGPRs (gp==0 waves) for the whole sequence.
  float c0v[8], c1v[8];
#pragma unroll
  for (int v = 0; v < 8; ++v) { c0v[v] = 0.f; c1v[v] = 0.f; }

  unsigned sync_no = 0;

  for (int t = 0; t < SRC_LEN + TGT_LEN; ++t) {
    const int pin = t & 1, pout = pin ^ 1;

    // ---------- layer 0: g = x_t*Wih0^T + h0*Whh0^T + b0 ----------
    {
      v8f acc[2] = {};
      const bf16_t* X; int ldx;
      if (t < SRC_LEN) {
        X   = srcb + ((size_t)rowbase * SRC_LEN + t) * IN_DIM;
        ldx = SRC_LEN * IN_DIM;
      } else {
        X   = tgtb + ((size_t)rowbase * TGT_LEN + (t - SRC_LEN)) * OUT_DIM;
        ldx = TGT_LEN * OUT_DIM;
      }
      // prefetch next timestep's x rows (global_prefetch_b8)
      if (t + 1 < SRC_LEN + TGT_LEN) {
        const bf16_t* Xn; int ldn;
        if (t + 1 < SRC_LEN) {
          Xn  = srcb + ((size_t)rowbase * SRC_LEN + (t + 1)) * IN_DIM;
          ldn = SRC_LEN * IN_DIM;
        } else {
          Xn  = tgtb + ((size_t)rowbase * TGT_LEN + (t + 1 - SRC_LEN)) * OUT_DIM;
          ldn = TGT_LEN * OUT_DIM;
        }
        __builtin_prefetch(Xn + (size_t)nn * ldn, 0, 1);
      }

      mma2_glb(acc, X, ldx,
               Wih0 + (size_t)(g0 * HID + c0) * IN_DIM,
               Wih0 + (size_t)(g1 * HID + c0) * IN_DIM, IN_DIM, IN_DIM, lane);
      mma2_lds(acc, h0buf + ((size_t)pin * BATCH + rowbase) * HID, HID,
               (unsigned)(SLAB0 + g0 * 16 * HID),
               (unsigned)(SLAB0 + g1 * 16 * HID), lane);

      if (gp == 1) {   // hand (g,o) pre-activation tiles to the (i,f) wave
        *(v8f*)&scratch[((rt * 2 + 0) * 32 + lane) * 8] = acc[0];
        *(v8f*)&scratch[((rt * 2 + 1) * 32 + lane) * 8] = acc[1];
      }
      __syncthreads();
      if (gp == 0) {
        v8f ga = *(const v8f*)&scratch[((rt * 2 + 0) * 32 + lane) * 8];
        v8f oa = *(const v8f*)&scratch[((rt * 2 + 1) * 32 + lane) * 8];
        bf16_t* Hn = h0buf + (size_t)pout * BATCH * HID;
#pragma unroll
        for (int v = 0; v < 8; ++v) {
          float iv = sigf (acc[0][v] + bias0[0]);
          float fv = sigf (acc[1][v] + bias0[1]);
          float gv = tanhf(ga[v]     + bias0[2]);
          float ov = sigf (oa[v]     + bias0[3]);
          float c  = fv * c0v[v] + iv * gv;
          c0v[v]   = c;
          float h  = ov * tanhf(c);
          Hn[(size_t)(rowbase + v + 8 * hi) * HID + c0 + nn] = (bf16_t)h;
        }
      }
    }
    grid_barrier(sync, NBLK * (++sync_no));

    // ---------- layer 1: g = h0_t*Wih1^T + h1*Whh1^T + b1 ----------
    {
      v8f acc[2] = {};
      mma2_lds(acc, h0buf + ((size_t)pout * BATCH + rowbase) * HID, HID,
               (unsigned)(SLAB1 + g0 * 16 * HID),
               (unsigned)(SLAB1 + g1 * 16 * HID), lane);
      mma2_glb(acc, h1buf + ((size_t)pin * BATCH + rowbase) * HID, HID,
               Whh1 + (size_t)(g0 * HID + c0) * HID,
               Whh1 + (size_t)(g1 * HID + c0) * HID, HID, HID, lane);

      if (gp == 1) {
        *(v8f*)&scratch[((rt * 2 + 0) * 32 + lane) * 8] = acc[0];
        *(v8f*)&scratch[((rt * 2 + 1) * 32 + lane) * 8] = acc[1];
      }
      __syncthreads();
      if (gp == 0) {
        v8f ga = *(const v8f*)&scratch[((rt * 2 + 0) * 32 + lane) * 8];
        v8f oa = *(const v8f*)&scratch[((rt * 2 + 1) * 32 + lane) * 8];
        bf16_t* Hn = h1buf + (size_t)pout * BATCH * HID;
#pragma unroll
        for (int v = 0; v < 8; ++v) {
          float iv = sigf (acc[0][v] + bias1[0]);
          float fv = sigf (acc[1][v] + bias1[1]);
          float gv = tanhf(ga[v]     + bias1[2]);
          float ov = sigf (oa[v]     + bias1[3]);
          float c  = fv * c1v[v] + iv * gv;
          c1v[v]   = c;
          float h  = ov * tanhf(c);
          Hn[(size_t)(rowbase + v + 8 * hi) * HID + c0 + nn] = (bf16_t)h;
        }
      }
    }
    grid_barrier(sync, NBLK * (++sync_no));

    // ---------- FC projection (decode steps only), 32 spare blocks ----------
    if (t >= SRC_LEN && blk < 32 && wave == 0) {
      const int rtc = blk >> 3, ct = blk & 7;
      v8f acc = {};
      const bf16_t* Hp = h1buf + ((size_t)pout * BATCH + rtc * 16) * HID;
      for (int k0 = 0; k0 < HID; k0 += 32) {
        v16bf a = load_a16(Hp + k0, HID, lane);
        v16bf b = load_b16(Wfc + (size_t)(ct * 16) * HID + k0, HID, lane);
        acc = wmma_bf16(a, b, acc);
      }
      const float bias = bfc[ct * 16 + nn];
      const int   tp   = t - SRC_LEN;
#pragma unroll
      for (int v = 0; v < 8; ++v) {
        const int brow = rtc * 16 + v + 8 * hi;
        out[((size_t)brow * TGT_LEN + tp) * OUT_DIM + ct * 16 + nn] = acc[v] + bias;
      }
    }
  }
}

extern "C" void kernel_launch(void* const* d_in, const int* in_sizes, int n_in,
                              void* d_out, int out_size, void* d_ws, size_t ws_size,
                              hipStream_t stream) {
  (void)in_sizes; (void)n_in; (void)out_size; (void)ws_size;
  const float* src  = (const float*)d_in[0];
  const float* tgt  = (const float*)d_in[1];
  const float* Wih0 = (const float*)d_in[2];
  const float* Whh0 = (const float*)d_in[3];
  const float* bih0 = (const float*)d_in[4];
  const float* bhh0 = (const float*)d_in[5];
  const float* Wih1 = (const float*)d_in[6];
  const float* Whh1 = (const float*)d_in[7];
  const float* bih1 = (const float*)d_in[8];
  const float* bhh1 = (const float*)d_in[9];
  const float* Wfc  = (const float*)d_in[10];
  const float* bfc  = (const float*)d_in[11];
  float* out = (float*)d_out;

  char*  ws  = (char*)d_ws;
  size_t off = 0;
  auto take = [&](size_t bytes) -> char* {
    char* p = ws + off;
    off = (off + bytes + 255) & ~(size_t)255;
    return p;
  };

  const size_t nWih0 = (size_t)4 * HID * IN_DIM;
  const size_t nWhh  = (size_t)4 * HID * HID;
  const size_t nWfc  = (size_t)OUT_DIM * HID;
  const size_t nSrc  = (size_t)BATCH * SRC_LEN * IN_DIM;
  const size_t nTgt  = (size_t)BATCH * TGT_LEN * OUT_DIM;
  const size_t nH    = (size_t)2 * BATCH * HID;

  bf16_t* wih0b = (bf16_t*)take(nWih0 * sizeof(bf16_t));
  bf16_t* whh0b = (bf16_t*)take(nWhh  * sizeof(bf16_t));
  bf16_t* wih1b = (bf16_t*)take(nWhh  * sizeof(bf16_t));
  bf16_t* whh1b = (bf16_t*)take(nWhh  * sizeof(bf16_t));
  bf16_t* wfcb  = (bf16_t*)take(nWfc  * sizeof(bf16_t));
  bf16_t* srcb  = (bf16_t*)take(nSrc  * sizeof(bf16_t));
  bf16_t* tgtb  = (bf16_t*)take(nTgt  * sizeof(bf16_t));
  float*  b0    = (float*)take(4 * HID * sizeof(float));
  float*  b1    = (float*)take(4 * HID * sizeof(float));
  bf16_t* h0buf = (bf16_t*)take(nH * sizeof(bf16_t));
  bf16_t* h1buf = (bf16_t*)take(nH * sizeof(bf16_t));
  unsigned* sync = (unsigned*)take(256);

  hipMemsetAsync(h0buf, 0, nH * sizeof(bf16_t), stream);
  hipMemsetAsync(h1buf, 0, nH * sizeof(bf16_t), stream);
  hipMemsetAsync(sync,  0, 256, stream);

  const dim3 cb(256), cg(2048);
  cvt_f32_bf16<<<cg, cb, 0, stream>>>(Wih0, wih0b, (long long)nWih0);
  cvt_f32_bf16<<<cg, cb, 0, stream>>>(Whh0, whh0b, (long long)nWhh);
  cvt_f32_bf16<<<cg, cb, 0, stream>>>(Wih1, wih1b, (long long)nWhh);
  cvt_f32_bf16<<<cg, cb, 0, stream>>>(Whh1, whh1b, (long long)nWhh);
  cvt_f32_bf16<<<cg, cb, 0, stream>>>(Wfc,  wfcb,  (long long)nWfc);
  cvt_f32_bf16<<<cg, cb, 0, stream>>>(src,  srcb,  (long long)nSrc);
  cvt_f32_bf16<<<cg, cb, 0, stream>>>(tgt,  tgtb,  (long long)nTgt);
  add_bias<<<(4 * HID + 255) / 256, 256, 0, stream>>>(bih0, bhh0, b0, 4 * HID);
  add_bias<<<(4 * HID + 255) / 256, 256, 0, stream>>>(bih1, bhh1, b1, 4 * HID);

  // 256 KB weight slabs + 8 KB f32 scratch (<= 320 KB WGP LDS)
  const int shmem = LDS_ELEMS * (int)sizeof(bf16_t) + SCRATCH_F32 * (int)sizeof(float);
  hipFuncSetAttribute((const void*)lstm_seq2seq,
                      hipFuncAttributeMaxDynamicSharedMemorySize, shmem);
  lstm_seq2seq<<<dim3(NBLK), dim3(NTHR), shmem, stream>>>(
      srcb, tgtb, wih0b, whh0b, wih1b, whh1b, wfcb,
      b0, b1, bfc, h0buf, h1buf, out, sync);
}
